// StateSpaceNet_2491081031711
// MI455X (gfx1250) — matrix operand order repaired and checked by reading, not compiled, not run on hardware
//
#include <hip/hip_runtime.h>
#include <hip/hip_bf16.h>

// Dims from the reference
#define IO    192
#define SD    192
#define CAT   384          // IO + SD
#define BATCH 32
#define TSTEPS 32
#define NW    36864        // rows per weight matrix (= 192*192)
#define NTOT  (4*NW)       // 147456 combined output columns [Wa;Wb;Wc;Wd]
#define RMS_EPS 1.1920929e-07f

typedef __attribute__((ext_vector_type(16))) __bf16 v16bf;
typedef __attribute__((ext_vector_type(8)))  float  v8f;

union FragBF { v16bf v; uint4 u[2]; };

__device__ __forceinline__ unsigned short f32_to_bf16(float f) {
    unsigned int u = __float_as_uint(f);
    unsigned int r = u + 0x7fffu + ((u >> 16) & 1u);   // round-to-nearest-even
    return (unsigned short)(r >> 16);
}

// ---------------------------------------------------------------------------
// 1) One-time weight conversion fp32 -> bf16, packed [Wa;Wb;Wc;Wd] (147456x384)
//    113.2 MB of bf16 weights -> fits in the 192 MB L2, so the 32-step loop
//    re-reads them at L2 bandwidth instead of HBM (fp32 would be 226 MB and
//    stream from HBM every step).
// ---------------------------------------------------------------------------
__global__ __launch_bounds__(256) void convert_weights(
    const float* __restrict__ Wa, const float* __restrict__ Wb,
    const float* __restrict__ Wc, const float* __restrict__ Wd,
    unsigned short* __restrict__ dst)
{
    int w = blockIdx.y;
    const float* src = (w == 0) ? Wa : (w == 1) ? Wb : (w == 2) ? Wc : Wd;
    size_t idx = ((size_t)blockIdx.x * blockDim.x + threadIdx.x) * 4;
    const size_t total = (size_t)NW * CAT;
    if (idx >= total) return;
    float4 f = *(const float4*)(src + idx);
    uint2 packed;
    packed.x = (unsigned)f32_to_bf16(f.x) | ((unsigned)f32_to_bf16(f.y) << 16);
    packed.y = (unsigned)f32_to_bf16(f.z) | ((unsigned)f32_to_bf16(f.w) << 16);
    *(uint2*)(dst + (size_t)w * total + idx) = packed;
}

// ---------------------------------------------------------------------------
// 2) Init: state = 0, c0 = [x(:,0,:), 0] in bf16
// ---------------------------------------------------------------------------
__global__ void init_state(const float* __restrict__ x,
                           unsigned short* __restrict__ cbuf,
                           float* __restrict__ state)
{
    int b = blockIdx.x, j = threadIdx.x;            // 192 threads
    cbuf[b * CAT + j]      = f32_to_bf16(x[(size_t)b * TSTEPS * IO + j]);
    cbuf[b * CAT + IO + j] = 0;
    state[b * SD + j]      = 0.0f;
}

// ---------------------------------------------------------------------------
// 3) Per-step GEMM: G[32][147456] = c[32][384] @ Wbf^T, -exp() fused on the
//    Wa block. Each wave owns a 32x64 output span: 4 N-tiles of 16 cols,
//    two accumulators each (M=0..15, M=16..31). Per K-chunk of 32:
//    A fragments loaded ONCE (4x b128) and reused across the 4 N-tiles
//    (8x b128 weight loads, 8 WMMAs) -> 1.5 loads per v_wmma instead of 3.
//    96 v_wmma_f32_16x16x32_bf16 per wave.
//    Fragment layouts per CDNA5 ISA:
//      B (32x16): lane l -> column n+(l&15), halves 0-7 = K kk+sel*8..,
//                 halves 8-15 = K kk+16+sel*8..
//      A (16x32): lane l -> row (l&15), halves 0-7 = K kk+sel*8..,
//                 halves 8-15 = K kk+16+sel*8..
// ---------------------------------------------------------------------------
__global__ __launch_bounds__(256) void gemm_step(
    const unsigned short* __restrict__ Wbf,   // [NTOT][CAT] bf16
    const unsigned short* __restrict__ Cbf,   // [BATCH][CAT] bf16
    float* __restrict__ G)                    // [BATCH][NTOT] f32
{
    const int lane  = threadIdx.x & 31;
    const int wave  = threadIdx.x >> 5;
    const int tile0 = (blockIdx.x * 8 + wave) * 4;  // 4 consecutive 16-col tiles
    const int n0    = tile0 * 16;                   // 64-column span
    const int sel   = lane >> 4;                    // 0 or 1
    const int nl    = n0 + (lane & 15);

    const unsigned short* wp  = Wbf + (size_t)nl * CAT + sel * 8;
    const unsigned short* ap0 = Cbf + (size_t)(lane & 15) * CAT + sel * 8;
    const unsigned short* ap1 = ap0 + (size_t)16 * CAT;

    v8f acc0[4] = {{0.f}, {0.f}, {0.f}, {0.f}};     // rows M = 0..15
    v8f acc1[4] = {{0.f}, {0.f}, {0.f}, {0.f}};     // rows M = 16..31

#pragma unroll
    for (int kk = 0; kk < CAT; kk += 32) {
        FragBF a0, a1;
        a0.u[0] = *(const uint4*)(ap0 + kk);
        a0.u[1] = *(const uint4*)(ap0 + kk + 16);
        a1.u[0] = *(const uint4*)(ap1 + kk);
        a1.u[1] = *(const uint4*)(ap1 + kk + 16);
#pragma unroll
        for (int u = 0; u < 4; ++u) {
            FragBF bf;
            const unsigned short* wpu = wp + (size_t)u * 16 * CAT;
            bf.u[0] = *(const uint4*)(wpu + kk);
            bf.u[1] = *(const uint4*)(wpu + kk + 16);
            acc0[u] = __builtin_amdgcn_wmma_f32_16x16x32_bf16(
                          false, a0.v, false, bf.v, (short)0, acc0[u], false, false);
            acc1[u] = __builtin_amdgcn_wmma_f32_16x16x32_bf16(
                          false, a1.v, false, bf.v, (short)0, acc1[u], false, false);
        }
    }

    // C/D layout: VGPR r, lanes 0-15 -> M=r, lanes 16-31 -> M=r+8, N = lane&15
    const bool isA  = (n0 < NW);   // 64-col span never straddles (36864 % 64 == 0)
    const int mbase = sel * 8;
#pragma unroll
    for (int u = 0; u < 4; ++u) {
        float* gp = G + nl + u * 16;               // column, row stride NTOT
#pragma unroll
        for (int r = 0; r < 8; ++r) {
            float v0 = acc0[u][r];
            float v1 = acc1[u][r];
            if (isA) { v0 = -__expf(v0); v1 = -__expf(v1); }
            gp[(size_t)(mbase + r)      * NTOT] = v0;
            gp[(size_t)(mbase + r + 16) * NTOT] = v1;
        }
    }
}

// ---------------------------------------------------------------------------
// 4) Per-step recurrence: one workgroup per batch, 192x3 threads.
//    new_state[t] = sum_s st[s]*A[s,t] + sum_d x[d]*B[d,t]
//    out[d]       = sum_s ns[s]*C[s,d] + sum_e x[e]*D[e,d]
//    y = rmsnorm(out + x) * w ; also emit next-step state and bf16 c.
//    Column reads of G are fully coalesced (192 consecutive floats per row).
// ---------------------------------------------------------------------------
__global__ __launch_bounds__(576) void update_step(
    const float* __restrict__ x, const float* __restrict__ G,
    const float* __restrict__ wn, float* __restrict__ state,
    unsigned short* __restrict__ cbuf, float* __restrict__ y, int t)
{
    __shared__ float s_st[SD], s_x[IO], s_ns[SD];
    __shared__ float part[3][SD];
    __shared__ float s_msq;

    const int b = blockIdx.x;
    const int j = threadIdx.x;            // 0..191  (output index)
    const int p = threadIdx.y;            // 0..2    (K-split)
    const float* Gr = G + (size_t)b * NTOT;

    if (p == 0) {
        s_st[j] = state[b * SD + j];
        s_x[j]  = x[(size_t)b * TSTEPS * IO + (size_t)t * IO + j];
    }
    __syncthreads();

    // phase 1: new_state
    {
        const float* Ap = Gr + j;              // A[s*192 + j]
        const float* Bp = Gr + NW + j;         // B[d*192 + j]
        float acc = 0.0f;
        const int s0 = p * 64;
        for (int s = s0; s < s0 + 64; ++s) acc += s_st[s] * Ap[s * SD];
        for (int d = s0; d < s0 + 64; ++d) acc += s_x[d]  * Bp[d * SD];
        part[p][j] = acc;
    }
    __syncthreads();
    if (p == 0) {
        float ns = part[0][j] + part[1][j] + part[2][j];
        s_ns[j] = ns;
        state[b * SD + j] = ns;                // next step's state
    }
    __syncthreads();

    // phase 2: out
    {
        const float* Cp = Gr + 2 * NW + j;     // C[s*192 + j]
        const float* Dp = Gr + 3 * NW + j;     // D[e*192 + j]
        float acc = 0.0f;
        const int s0 = p * 64;
        for (int s = s0; s < s0 + 64; ++s) acc += s_ns[s] * Cp[s * IO];
        for (int d = s0; d < s0 + 64; ++d) acc += s_x[d]  * Dp[d * IO];
        part[p][j] = acc;
    }
    __syncthreads();
    if (p == 0) {
        float h = part[0][j] + part[1][j] + part[2][j] + s_x[j];
        part[1][j] = h * h;                    // reduction buffer
        part[2][j] = h;                        // stash h
    }
    __syncthreads();
    for (int stp = 96; stp >= 3; stp >>= 1) {  // 192 -> 3
        if (p == 0 && j < stp) part[1][j] += part[1][j + stp];
        __syncthreads();
    }
    if (p == 0 && j == 0)
        s_msq = (part[1][0] + part[1][1] + part[1][2]) * (1.0f / (float)IO);
    __syncthreads();

    if (p == 0) {
        float h = part[2][j];
        y[(size_t)b * TSTEPS * IO + (size_t)t * IO + j] =
            h * rsqrtf(s_msq + RMS_EPS) * wn[j];
        // next-step c = [x_{t+1}, new_state] in bf16
        cbuf[b * CAT + IO + j] = f32_to_bf16(s_ns[j]);
        if (t + 1 < TSTEPS)
            cbuf[b * CAT + j] =
                f32_to_bf16(x[(size_t)b * TSTEPS * IO + (size_t)(t + 1) * IO + j]);
    }
}

// ---------------------------------------------------------------------------
extern "C" void kernel_launch(void* const* d_in, const int* in_sizes, int n_in,
                              void* d_out, int out_size, void* d_ws, size_t ws_size,
                              hipStream_t stream) {
    const float* x  = (const float*)d_in[0];
    const float* Wa = (const float*)d_in[1];
    const float* Wb = (const float*)d_in[2];
    const float* Wc = (const float*)d_in[3];
    const float* Wd = (const float*)d_in[4];
    const float* wn = (const float*)d_in[5];

    // workspace layout (needs ~132.2 MB)
    char* ws = (char*)d_ws;
    unsigned short* wbf = (unsigned short*)ws;                 // 113,246,208 B
    size_t off = (size_t)NTOT * CAT * sizeof(unsigned short);
    float* G = (float*)(ws + off);                             // 18,874,368 B
    off += (size_t)BATCH * NTOT * sizeof(float);
    float* state = (float*)(ws + off);                         // 24,576 B
    off += (size_t)BATCH * SD * sizeof(float);
    unsigned short* cbuf = (unsigned short*)(ws + off);        // 24,576 B

    float* y = (float*)d_out;

    // 1) fp32 -> bf16 weights (once per launch; deterministic)
    convert_weights<<<dim3((NW * CAT / 4 + 255) / 256, 4), 256, 0, stream>>>(
        Wa, Wb, Wc, Wd, wbf);

    // 2) state = 0, c0 = [x0, 0]
    init_state<<<BATCH, IO, 0, stream>>>(x, cbuf, state);

    // 3) sequential recurrence: 9216 tiles = 288 blocks x 8 waves x 4 tiles
    for (int t = 0; t < TSTEPS; ++t) {
        gemm_step<<<NTOT / 16 / 8 / 4, 256, 0, stream>>>(wbf, cbuf, G);
        update_step<<<BATCH, dim3(IO, 3), 0, stream>>>(x, G, wn, state, cbuf, y, t);
    }
}
